// GraphNetwork_82660940579185
// MI455X (gfx1250) — compile-verified, matrix operand order checked by reading, not run on hardware
//
#include <hip/hip_runtime.h>
#include <hip/hip_bf16.h>
#include <cstdint>

typedef _Float16 f16;
typedef __attribute__((ext_vector_type(16))) _Float16 v16h;
typedef __attribute__((ext_vector_type(8)))  _Float16 v8h;
typedef __attribute__((ext_vector_type(4)))  _Float16 v4h;
typedef __attribute__((ext_vector_type(8)))  float    v8f;
typedef __attribute__((ext_vector_type(4)))  float    v4f;

#define NNODES 98304   // T*NX*NX = 6*128*128
#define BATCH  4
#define MROWS  (NNODES * BATCH)  // 393216 GEMM rows (= 1536 * 256)

// ---------------------------------------------------------------------------
// Gather + pack: X[r, 0:F) = f16(h[r]) ; X[r, F:2F) = f16(mean of 4 nbrs).
// Fuses the self/neighbor GEMM pair into one K-concatenated GEMM operand.
// SRC_BNF=true reads the original [B, N, F] input layout; otherwise [N*B, F]
// rows (r = n*B + b) with row stride / column offset (for the U-net concats).
// Grid is exact (MROWS*F/4 divisible by 256) -> no bounds guard.
// ---------------------------------------------------------------------------
template <int F, bool SRC_BNF, int SRCSTRIDE, int COFF>
__global__ __launch_bounds__(256) void gather_pack(
    const float* __restrict__ src, const int* __restrict__ nbr,
    f16* __restrict__ X) {
  constexpr int F4 = F / 4;
  const unsigned id = blockIdx.x * 256u + threadIdx.x;
  const unsigned f  = (id % F4) * 4u;
  const unsigned rb = id / F4;
  const unsigned n  = rb / BATCH;
  const unsigned b  = rb % BATCH;

  const int4 nn = *(const int4*)(nbr + n * 4u);

  auto base = [&](unsigned node) -> unsigned {  // all buffers < 2^31 elements
    return SRC_BNF ? (b * (unsigned)NNODES + node) * SRCSTRIDE + COFF + f
                   : (node * (unsigned)BATCH + b) * SRCSTRIDE + COFF + f;
  };

  const v4f s  = *(const v4f*)(src + base(n));
  const v4f m0 = *(const v4f*)(src + base((unsigned)nn.x));
  const v4f m1 = *(const v4f*)(src + base((unsigned)nn.y));
  const v4f m2 = *(const v4f*)(src + base((unsigned)nn.z));
  const v4f m3 = *(const v4f*)(src + base((unsigned)nn.w));
  const v4f m  = (m0 + m1 + m2 + m3) * 0.25f;

  f16* xr = X + rb * (2u * F);
  v4h hs = {(f16)s.x, (f16)s.y, (f16)s.z, (f16)s.w};
  v4h hm = {(f16)m.x, (f16)m.y, (f16)m.z, (f16)m.w};
  *(v4h*)(xr + f)     = hs;
  *(v4h*)(xr + F + f) = hm;
}

// ---------------------------------------------------------------------------
// Pack [Ws; Wn] (each f32 [din, dout]) into f16 Wt[dout][2*din] (row = out col)
// so B-fragment loads are contiguous along K, matching the A-fragment code.
// All totals are multiples of 256 -> no bounds guard.
// ---------------------------------------------------------------------------
__global__ __launch_bounds__(256) void pack_weights(
    const float* __restrict__ Ws, const float* __restrict__ Wn,
    f16* __restrict__ Wt, int din, int dout) {
  const int K2 = 2 * din;
  const int id = blockIdx.x * 256 + threadIdx.x;
  const int o = id / K2;
  const int k = id % K2;
  const float v = (k < din) ? Ws[k * dout + o] : Wn[(k - din) * dout + o];
  Wt[o * K2 + k] = (f16)v;
}

// ---------------------------------------------------------------------------
// Fused dual-GEMM + bias + optional ReLU:  out = act(X[M,K2] @ Wt^T + b)
// Block: 256 threads (8 wave32), 64 rows of X staged in LDS, each wave owns
// a statically-unrolled set of 16x16 tiles (TILES/8 per wave -> uniform
// control flow, EXEC all-ones at every WMMA), v_wmma_f32_16x16x32_f16 with
// f32 accumulators, compile-time output stride/offset (32-bit addressing).
// Fragment layouts per CDNA5 ISA 7.12.2 (16-bit A 16x32 / B 32x16 / f32 C).
// ---------------------------------------------------------------------------
template <int K2, int NOUT, bool RELU, bool TRANSPOSE_OUT, int OSTRIDE, int COFF>
__global__ __launch_bounds__(256) void sage_gemm(
    const f16* __restrict__ X, const f16* __restrict__ Wt,
    const float* __restrict__ bias, float* __restrict__ out) {
  __shared__ f16 Xs[64 * K2];
  const unsigned tid  = threadIdx.x;
  const unsigned row0 = blockIdx.x * 64u;

  // Prefetch the next block's X region (global_prefetch_b8), one cacheline
  // stride per thread; speculative, OOB addresses are silently dropped.
  __builtin_prefetch(X + (size_t)(row0 + 64) * K2 + tid * (K2 / 4), 0, 1);

  // Cooperative 16B-chunk stage of 64 contiguous X rows into LDS.
  {
    const uint4* s = (const uint4*)(X + row0 * (unsigned)K2);
    uint4* d = (uint4*)Xs;
    constexpr int chunks = 64 * K2 / 8;
#pragma unroll
    for (int i = tid; i < chunks; i += 256) d[i] = s[i];
  }
  __syncthreads();

  const unsigned wave  = tid >> 5;
  const unsigned lane  = tid & 31;
  const unsigned lanlo = lane & 15;
  const unsigned khalf = (lane >> 4) * 8;  // K sub-slice: 0..7/16..23 vs 8..15/24..31
  constexpr int TN = NOUT / 16;
  constexpr int TILES = 4 * TN;   // 8, 16 or 32 -> exactly TILES/8 per wave
  constexpr int ITERS = TILES / 8;

#pragma unroll
  for (int it = 0; it < ITERS; ++it) {
    const unsigned t   = wave + it * 8u;
    const unsigned mt  = t / TN;
    const unsigned nt  = t % TN;
    const unsigned col = nt * 16u + lanlo;

    const float bv = bias[col];
    v8f c = {bv, bv, bv, bv, bv, bv, bv, bv};

    const f16* xrow = Xs + (mt * 16u + lanlo) * K2 + khalf;
    const f16* wrow = Wt + col * (unsigned)K2 + khalf;
#pragma unroll
    for (int k0 = 0; k0 < K2; k0 += 32) {
      v8h alo = *(const v8h*)(xrow + k0);
      v8h ahi = *(const v8h*)(xrow + k0 + 16);
      v8h blo = *(const v8h*)(wrow + k0);
      v8h bhi = *(const v8h*)(wrow + k0 + 16);
      v16h a  = __builtin_shufflevector(alo, ahi, 0, 1, 2, 3, 4, 5, 6, 7,
                                        8, 9, 10, 11, 12, 13, 14, 15);
      v16h bm = __builtin_shufflevector(blo, bhi, 0, 1, 2, 3, 4, 5, 6, 7,
                                        8, 9, 10, 11, 12, 13, 14, 15);
      c = __builtin_amdgcn_wmma_f32_16x16x32_f16(false, a, false, bm,
                                                 (short)0, c, false, false);
    }

    // C/D layout: VGPR r -> row r (+8 for high lane half), col = lane&15.
    const unsigned rbase = row0 + mt * 16u + ((lane >> 4) * 8u);
#pragma unroll
    for (int r = 0; r < 8; ++r) {
      float v = c[r];
      if (RELU) v = v > 0.f ? v : 0.f;
      const unsigned rg = rbase + r;
      if (TRANSPOSE_OUT) {  // final layer: write [B, N, NOUT] directly
        out[(rg & 3u) * (unsigned)(NNODES * NOUT) + (rg >> 2) * NOUT + col] = v;
      } else {
        out[rg * (unsigned)OSTRIDE + COFF + col] = v;
      }
    }
  }
}

// ---------------------------------------------------------------------------
extern "C" void kernel_launch(void* const* d_in, const int* in_sizes, int n_in,
                              void* d_out, int out_size, void* d_ws, size_t ws_size,
                              hipStream_t stream) {
  (void)in_sizes; (void)n_in; (void)out_size; (void)ws_size;
  const float* inp = (const float*)d_in[0];
  const int*   nbr = (const int*)d_in[1];
  const float *Ws[6], *Wn[6], *bs[6];
  for (int l = 0; l < 6; ++l) {
    Ws[l] = (const float*)d_in[2 + 3 * l];
    Wn[l] = (const float*)d_in[3 + 3 * l];
    bs[l] = (const float*)d_in[4 + 3 * l];
  }

  char* ws = (char*)d_ws;
  const size_t M = MROWS;
  f16* wt[6];
  for (int l = 0; l < 6; ++l) wt[l] = (f16*)(ws + (size_t)l * 65536);
  size_t off = 6 * 65536;
  f16*   Xp    = (f16*)(ws + off);   off += M * 256 * sizeof(f16);   // packed [h|m]
  float* h1    = (float*)(ws + off); off += M * 128 * sizeof(float); // [M,128]
  float* h6buf = (float*)(ws + off); off += M * 128 * sizeof(float); // cols 64:128=h2
  float* h5buf = (float*)(ws + off); off += M * 64  * sizeof(float); // cols 32:64=h3
  float* h4    = (float*)(ws + off); off += M * 32  * sizeof(float);
  float* outp  = (float*)d_out;

  const int din[6]  = {64, 128, 64, 32, 64, 128};
  const int dout[6] = {128, 64, 32, 32, 64, 64};
  for (int l = 0; l < 6; ++l) {
    const int total = dout[l] * 2 * din[l];  // all multiples of 256
    pack_weights<<<total / 256, 256, 0, stream>>>(Ws[l], Wn[l], wt[l],
                                                  din[l], dout[l]);
  }

  const dim3 blk(256);
  auto gblocks = [](int F) { return dim3((unsigned)(MROWS * (F / 4) / 256)); };
  const int GG = MROWS / 64;

  // L1: x(BNF,64) -> h1[M,128]
  gather_pack<64, true, 64, 0><<<gblocks(64), blk, 0, stream>>>(inp, nbr, Xp);
  sage_gemm<128, 128, true, false, 128, 0><<<GG, blk, 0, stream>>>(Xp, wt[0], bs[0], h1);
  // L2: h1 -> h2 (cols 64:128 of h6buf)
  gather_pack<128, false, 128, 0><<<gblocks(128), blk, 0, stream>>>(h1, nbr, Xp);
  sage_gemm<256, 64, true, false, 128, 64><<<GG, blk, 0, stream>>>(Xp, wt[1], bs[1], h6buf);
  // L3: h2 -> h3 (cols 32:64 of h5buf)
  gather_pack<64, false, 128, 64><<<gblocks(64), blk, 0, stream>>>(h6buf, nbr, Xp);
  sage_gemm<128, 32, true, false, 64, 32><<<GG, blk, 0, stream>>>(Xp, wt[2], bs[2], h5buf);
  // L4: h3 -> h4[M,32]
  gather_pack<32, false, 64, 32><<<gblocks(32), blk, 0, stream>>>(h5buf, nbr, Xp);
  sage_gemm<64, 32, true, false, 32, 0><<<GG, blk, 0, stream>>>(Xp, wt[3], bs[3], h4);
  // L5a: act(sage(h4, W4)) -> cols 0:32 of h5buf  (h5 complete)
  gather_pack<32, false, 32, 0><<<gblocks(32), blk, 0, stream>>>(h4, nbr, Xp);
  sage_gemm<64, 32, true, false, 64, 0><<<GG, blk, 0, stream>>>(Xp, wt[3], bs[3], h5buf);
  // L5b: act(sage(h5, W5)) -> cols 0:64 of h6buf  (h6 complete)
  gather_pack<64, false, 64, 0><<<gblocks(64), blk, 0, stream>>>(h5buf, nbr, Xp);
  sage_gemm<128, 64, true, false, 128, 0><<<GG, blk, 0, stream>>>(Xp, wt[4], bs[4], h6buf);
  // L6: sage(h6, W6) -> out [B,N,64], no relu
  gather_pack<128, false, 128, 0><<<gblocks(128), blk, 0, stream>>>(h6buf, nbr, Xp);
  sage_gemm<256, 64, false, true, 64, 0><<<GG, blk, 0, stream>>>(Xp, wt[5], bs[5], outp);
}
